// MIMICDecayModel_40072044872048
// MI455X (gfx1250) — matrix operand order repaired and checked by reading, not compiled, not run on hardware
//
#include <hip/hip_runtime.h>

// ---------------------------------------------------------------------------
// MIMIC RIMs/GRU-D recurrent model on MI455X (gfx1250).
// 128 blocks x (16-row batch tile); block = 192 threads = 6 waves, one per rim.
// All GEMMs on v_wmma_f32_16x16x32_bf16 (bf16 in, f32 accumulate).
// B operands: pre-packed per-lane fragments in d_ws (32B contiguous per lane).
// A operands: bf16 LDS mirrors kept in the same packed layout -> one 32B
//             contiguous LDS read per fragment (2x ds_load_b128).
// This round: cap unrolling to kill scratch spills + shrink static code.
// ---------------------------------------------------------------------------

typedef __attribute__((ext_vector_type(16))) __bf16 v16bf;
typedef __attribute__((ext_vector_type(8)))  float  v8f;

// model dims
#define BB    2048
#define TT    50
#define FF    64
#define RR    6
#define HH    64
#define KSd   64
#define VSd   128
#define NCd   4
#define QCd   32
#define VCd   16
#define SSd   16
#define KACT  4
#define G3    192          // 3*H

#define NTHREADS 192       // 6 waves

// packed weight offsets (in bf16 elements) inside d_ws
#define OFF_WK   0                       // 1 mat, K=64,  N=64  -> 4096
#define OFF_WV   4096                    // 1 mat, K=64,  N=128 -> 8192
#define OFF_WQ   12288                   // 6 mat, K=64,  N=64  -> 24576
#define OFF_WIH  36864                   // 6 mat, K=128, N=192 -> 147456
#define OFF_WHH  184320                  // 6 mat, K=64,  N=192 -> 73728
#define OFF_WQC  258048                  // 6 mat, K=64,  N=128 -> 49152
#define OFF_WKC  307200                  // 6 mat, K=64,  N=128 -> 49152
#define OFF_WVC  356352                  // 6 mat, K=64,  N=64  -> 24576

// ---------------- helpers ----------------

static __device__ __forceinline__ __bf16 f2bf(float f) { return (__bf16)f; }
static __device__ __forceinline__ float  bf2f(__bf16 b) { return (float)b; }
static __device__ __forceinline__ float sigmoidf(float x) {
  return 1.0f / (1.0f + __expf(-x));
}

// v_wmma_f32_16x16x32_bf16 : D = A(16x32 bf16) * B(32x16 bf16) + C(16x16 f32)
static __device__ __forceinline__ v8f wmma_bf(v16bf a, v16bf b, v8f c) {
  return __builtin_amdgcn_wmma_f32_16x16x32_bf16(false, a, false, b,
                                                 (short)0, c, false, false);
}

// A-fragment packed layout for a [16][K] bf16 tile (ISA 7.12.2 16-bit A):
//   frag kt holds K [kt*32, kt*32+32); within a frag, lane l / half h maps to
//   m = l&15, k = kt*32 + 8*(l>=16) + 16*(h>=8) + (h&7).
static __device__ __forceinline__ int apack_idx(int m, int k) {
  int kt = k >> 5, kk = k & 31;
  int lane = (m & 15) + (((kk >> 3) & 1) << 4);
  int h = ((kk >> 4) << 3) + (kk & 7);
  return (kt << 9) + (lane << 4) + h;
}
// A fragment: one contiguous 32B LDS read per lane.
static __device__ __forceinline__ v16bf load_apk(const __bf16* base, int ktile,
                                                 int lane) {
  return *(const v16bf*)(base + ((size_t)ktile << 9) + ((size_t)lane << 4));
}
// B fragment: pre-packed lane-major (512 bf16 per 32x16 tile) -> 32B read.
static __device__ __forceinline__ v16bf load_b(const __bf16* w, int frag, int lane) {
  return *(const v16bf*)(w + ((size_t)frag << 9) + ((size_t)lane << 4));
}
// D fragment (f32 16x16) scatter to LDS [16][ldd] at column n0.
// Layout: vgpr v -> row v + 8*(lane>=16), col = lane&15.
static __device__ __forceinline__ void store_d_f32(float* p, int ldd, int n0,
                                                   int lane, v8f d) {
  int n  = n0 + (lane & 15);
  int mb = (lane >> 4) << 3;
#pragma unroll
  for (int v = 0; v < 8; ++v) p[(mb + v) * ldd + n] = d[v];
}
static __device__ __forceinline__ void store_d_bf(__bf16* p, int ldd, int n0,
                                                  int lane, v8f d) {
  int n  = n0 + (lane & 15);
  int mb = (lane >> 4) << 3;
#pragma unroll
  for (int v = 0; v < 8; ++v) p[(mb + v) * ldd + n] = f2bf(d[v]);
}

// ---------------- weight packing kernel ----------------
// src: nmat contiguous row-major f32 matrices [K][N]; dst: bf16 fragments.
// frag order within a matrix: frag = ntile*ktiles + ktile (K innermost).
// dst[lane*16+h] = W[ktile*32 + 8*(lane>=16) + 16*(h>=8) + (h&7)][ntile*16 + (lane&15)]
__global__ void pack_weights(const float* __restrict__ src,
                             __bf16* __restrict__ dst,
                             int nmat, int K, int N) {
  int ktiles = K >> 5, ntiles = N >> 4;
  long long total = (long long)nmat * ktiles * ntiles * 512;
  for (long long i = (long long)blockIdx.x * blockDim.x + threadIdx.x;
       i < total; i += (long long)gridDim.x * blockDim.x) {
    int h    = (int)(i & 15);
    int lane = (int)((i >> 4) & 31);
    long long fi = i >> 9;
    int ktile = (int)(fi % ktiles);
    long long t2 = fi / ktiles;
    int ntile = (int)(t2 % ntiles);
    int mat   = (int)(t2 / ntiles);
    int k = ktile * 32 + ((lane >> 4) << 3) + ((h >> 3) << 4) + (h & 7);
    int n = ntile * 16 + (lane & 15);
    dst[i] = f2bf(src[((long long)mat * K + k) * N + n]);
  }
}

// ---------------- main recurrent kernel ----------------

#define SM_F32_WORDS (6144*4 + 1024 + 2048 + 128 + 128)        // 27904 floats
#define SM_BF_HALVES (1024 + 12288*3 + 6144*2)                 // 50176 halves
#define SMEM_BYTES   (SM_F32_WORDS*4 + SM_BF_HALVES*2)         // 211968 B

__global__ void __launch_bounds__(NTHREADS)
rims_recurrence_kernel(const float* __restrict__ x,
                       const float* __restrict__ statics,
                       const float* __restrict__ maskp,
                       const float* __restrict__ delta,
                       const float* __restrict__ xlast,
                       const float* __restrict__ xmean,
                       const float* __restrict__ wdg,
                       const float* __restrict__ bdg,
                       const float* __restrict__ bih,
                       const float* __restrict__ bhh,
                       const float* __restrict__ W1,
                       const float* __restrict__ b1,
                       const float* __restrict__ W2,
                       const float* __restrict__ b2,
                       const __bf16* __restrict__ ws,
                       float* __restrict__ out) {
  extern __shared__ __align__(16) char smem[];
  float* sh_h  = (float*)smem;          // [R][16][H]   state (f32)
  float* sh_hn = sh_h  + 6144;          // [R][16][H]   candidate state
  float* sh_q  = sh_hn + 6144;          // [R][16][64]  Q staging / comm output
  float* sh_Km = sh_q  + 6144;          // [16][KS]
  float* sh_Vm = sh_Km + 1024;          // [16][VS]
  float* sh_vc = sh_Vm + 2048;          // [R][16][NC*VC]
  float* sh_sc = sh_vc + 6144;          // [16][8] attention scores per rim
  float* sh_am = sh_sc + 128;           // [16][8] active-rim mask
  __bf16* sh_xh = (__bf16*)(sh_am + 128);   // [16][F]      A-packed
  __bf16* sh_in = sh_xh + 1024;             // [R][16][VS]  A-packed
  __bf16* sh_qc = sh_in + 12288;            // [R][16][NC*QC] row-major
  __bf16* sh_kc = sh_qc + 12288;            // [R][16][NC*QC] row-major
  __bf16* sh_hb  = sh_kc + 12288;           // [R][16][H]   A-packed mirror of h
  __bf16* sh_hnb = sh_hb + 6144;            // [R][16][H]   A-packed mirror of h_new

  const __bf16* pk_wk  = ws + OFF_WK;
  const __bf16* pk_wv  = ws + OFF_WV;
  const __bf16* pk_wq  = ws + OFF_WQ;
  const __bf16* pk_wih = ws + OFF_WIH;
  const __bf16* pk_whh = ws + OFF_WHH;
  const __bf16* pk_wqc = ws + OFF_WQC;
  const __bf16* pk_wkc = ws + OFF_WKC;
  const __bf16* pk_wvc = ws + OFF_WVC;

  const int tid  = threadIdx.x;
  const int wave = tid >> 5;       // rim id
  const int lane = tid & 31;
  const int b0   = blockIdx.x * 16;

  const float inv_sk = 0.125f;                 // 1/sqrt(64)
  const float inv_sq = 0.17677669529663689f;   // 1/sqrt(32)

  // zero state (f32 + bf16 mirror)
  for (int i = tid; i < 6144; i += NTHREADS) sh_h[i] = 0.0f;
  for (int i = tid; i < 6144; i += NTHREADS) sh_hb[i] = f2bf(0.0f);
  __syncthreads();

#pragma unroll 1
  for (int t = 0; t < TT; ++t) {
    // ---- A: GRU-D imputation -> xh (bf16, A-packed) --------------------
#pragma unroll 1
    for (int i = tid; i < 16 * FF; i += NTHREADS) {
      int row = i >> 6, f = i & 63;
      int b = b0 + row;
      size_t idx = ((size_t)b * TT + t) * FF + f;
      float g  = __expf(-fmaxf(delta[idx] * wdg[f] + bdg[f], 0.0f));
      float mt = maskp[idx];
      float xh = mt * x[idx] +
                 (1.0f - mt) * (g * xlast[idx] + (1.0f - g) * xmean[b * FF + f]);
      sh_xh[apack_idx(row, f)] = f2bf(xh);
      if (t + 1 < TT) {   // prefetch next step's rows (global_prefetch_b8)
        __builtin_prefetch(&x[idx + FF], 0, 1);
        __builtin_prefetch(&delta[idx + FF], 0, 1);
        __builtin_prefetch(&maskp[idx + FF], 0, 1);
        __builtin_prefetch(&xlast[idx + FF], 0, 1);
      }
    }
    __syncthreads();

    // ---- B: Q_r = h_r @ Wq_r (all waves); Km (wave0); Vm (wave1) -------
    {
      const __bf16* Wq_r = pk_wq + (size_t)wave * (2 * 4 * 512);
      v16bf ah0 = load_apk(sh_hb + wave * 1024, 0, lane);
      v16bf ah1 = load_apk(sh_hb + wave * 1024, 1, lane);
#pragma unroll 2
      for (int nt = 0; nt < 4; ++nt) {
        v8f acc = {};
        acc = wmma_bf(ah0, load_b(Wq_r, nt * 2 + 0, lane), acc);
        acc = wmma_bf(ah1, load_b(Wq_r, nt * 2 + 1, lane), acc);
        store_d_f32(sh_q + wave * 1024, 64, nt * 16, lane, acc);
      }
    }
    if (wave == 0) {          // Km = xh @ Wk  [16][64]
      v16bf ax0 = load_apk(sh_xh, 0, lane);
      v16bf ax1 = load_apk(sh_xh, 1, lane);
#pragma unroll 2
      for (int nt = 0; nt < 4; ++nt) {
        v8f acc = {};
        acc = wmma_bf(ax0, load_b(pk_wk, nt * 2 + 0, lane), acc);
        acc = wmma_bf(ax1, load_b(pk_wk, nt * 2 + 1, lane), acc);
        store_d_f32(sh_Km, KSd, nt * 16, lane, acc);
      }
    }
    if (wave == 1) {          // Vm = xh @ Wv  [16][128]
      v16bf ax0 = load_apk(sh_xh, 0, lane);
      v16bf ax1 = load_apk(sh_xh, 1, lane);
#pragma unroll 2
      for (int nt = 0; nt < 8; ++nt) {
        v8f acc = {};
        acc = wmma_bf(ax0, load_b(pk_wv, nt * 2 + 0, lane), acc);
        acc = wmma_bf(ax1, load_b(pk_wv, nt * 2 + 1, lane), acc);
        store_d_f32(sh_Vm, VSd, nt * 16, lane, acc);
      }
    }
    __syncthreads();

    // ---- C: score_r[row] = inv_sk * <Q_r[row], Km[row]> ----------------
    if (lane < 16) {
      int row = lane;
      float s = 0.0f;
#pragma unroll 8
      for (int k = 0; k < KSd; ++k)
        s += sh_q[wave * 1024 + row * 64 + k] * sh_Km[row * KSd + k];
      sh_sc[row * 8 + wave] = s * inv_sk;   // softmax([s,0])[0] = sigmoid(s)
    }
    __syncthreads();

    // ---- D: top-K_ACT rims (largest score == least null attention) ----
    if (tid < 16) {
      int row = tid;
#pragma unroll 1
      for (int r = 0; r < RR; ++r) {
        float sr = sh_sc[row * 8 + r];
        int cnt = 0;
#pragma unroll
        for (int r2 = 0; r2 < RR; ++r2) {
          float s2 = sh_sc[row * 8 + r2];
          if (s2 > sr || (s2 == sr && r2 < r)) ++cnt;
        }
        sh_am[row * 8 + r] = (cnt < KACT) ? 1.0f : 0.0f;
      }
    }
    __syncthreads();

    // ---- E: rim inputs, GRU gates (WMMA), nonlinearity -> h_new --------
    {
      const int r = wave;
#pragma unroll 1
      for (int i = lane; i < 16 * VSd; i += 32) {
        int row = i >> 7, col = i & 127;
        float g = sigmoidf(sh_sc[row * 8 + r]);
        sh_in[r * 2048 + apack_idx(row, col)] = f2bf(g * sh_Vm[i]);
      }
      const __bf16* Wih = pk_wih + (size_t)r * (4 * 12 * 512);
      const __bf16* Whh = pk_whh + (size_t)r * (2 * 12 * 512);
      // hoisted A fragments: inputs (K=128 -> 4 frags), h (K=64 -> 2 frags)
      v16bf ai[4], ah[2];
#pragma unroll
      for (int ks = 0; ks < 4; ++ks) ai[ks] = load_apk(sh_in + r * 2048, ks, lane);
#pragma unroll
      for (int ks = 0; ks < 2; ++ks) ah[ks] = load_apk(sh_hb + r * 1024, ks, lane);
#pragma unroll 1
      for (int j = 0; j < 4; ++j) {      // 16-col group within each gate
        v8f air = {}, aiz = {}, ain = {}, ahr = {}, ahz = {}, ahn = {};
#pragma unroll 2
        for (int ks = 0; ks < 4; ++ks) { // K = VS = 128 (cap B-frag liveness)
          air = wmma_bf(ai[ks], load_b(Wih, (0 + j) * 4 + ks, lane), air);
          aiz = wmma_bf(ai[ks], load_b(Wih, (4 + j) * 4 + ks, lane), aiz);
          ain = wmma_bf(ai[ks], load_b(Wih, (8 + j) * 4 + ks, lane), ain);
        }
#pragma unroll
        for (int ks = 0; ks < 2; ++ks) { // K = H = 64
          ahr = wmma_bf(ah[ks], load_b(Whh, (0 + j) * 2 + ks, lane), ahr);
          ahz = wmma_bf(ah[ks], load_b(Whh, (4 + j) * 2 + ks, lane), ahz);
          ahn = wmma_bf(ah[ks], load_b(Whh, (8 + j) * 2 + ks, lane), ahn);
        }
        int n = lane & 15, mb = (lane >> 4) << 3;
        int col = j * 16 + n;
        float bir = bih[r * G3 + col],        bhr = bhh[r * G3 + col];
        float biz = bih[r * G3 + 64 + col],   bhz = bhh[r * G3 + 64 + col];
        float bin = bih[r * G3 + 128 + col],  bhn = bhh[r * G3 + 128 + col];
#pragma unroll
        for (int v = 0; v < 8; ++v) {
          int m = mb + v;
          float rg = sigmoidf(air[v] + bir + ahr[v] + bhr);
          float zg = sigmoidf(aiz[v] + biz + ahz[v] + bhz);
          float ng = tanhf(ain[v] + bin + rg * (ahn[v] + bhn));
          float ho = sh_h[r * 1024 + m * HH + col];
          float hv = (1.0f - zg) * ng + zg * ho;
          sh_hn[r * 1024 + m * HH + col] = hv;
          sh_hnb[r * 1024 + apack_idx(m, col)] = f2bf(hv);
        }
      }

      // ---- F: comm projections q,k (bf16), v (f32) from h_new_r -------
      const __bf16* Wqc = pk_wqc + (size_t)r * (2 * 8 * 512);
      const __bf16* Wkc = pk_wkc + (size_t)r * (2 * 8 * 512);
      const __bf16* Wvc = pk_wvc + (size_t)r * (2 * 4 * 512);
      v16bf an0 = load_apk(sh_hnb + r * 1024, 0, lane);
      v16bf an1 = load_apk(sh_hnb + r * 1024, 1, lane);
#pragma unroll 1
      for (int nt = 0; nt < 8; ++nt) {
        v8f aq = {}, ak = {};
        aq = wmma_bf(an0, load_b(Wqc, nt * 2 + 0, lane), aq);
        aq = wmma_bf(an1, load_b(Wqc, nt * 2 + 1, lane), aq);
        ak = wmma_bf(an0, load_b(Wkc, nt * 2 + 0, lane), ak);
        ak = wmma_bf(an1, load_b(Wkc, nt * 2 + 1, lane), ak);
        store_d_bf(sh_qc + r * 2048, NCd * QCd, nt * 16, lane, aq);
        store_d_bf(sh_kc + r * 2048, NCd * QCd, nt * 16, lane, ak);
        if (nt < 4) {
          v8f av = {};
          av = wmma_bf(an0, load_b(Wvc, nt * 2 + 0, lane), av);
          av = wmma_bf(an1, load_b(Wvc, nt * 2 + 1, lane), av);
          store_d_f32(sh_vc + r * 1024, NCd * VCd, nt * 16, lane, av);
        }
      }
    }
    __syncthreads();

    // ---- G: R x R communication attention (tiny; VALU) -----------------
    if (tid < 64) {
      int row = tid & 15, head = tid >> 4;
      const int qb = head * QCd, vb = head * VCd;
      float sc[RR][RR];
#pragma unroll 1
      for (int rq = 0; rq < RR; ++rq)
#pragma unroll 1
        for (int rs = 0; rs < RR; ++rs) {
          float s = 0.0f;
#pragma unroll
          for (int k = 0; k < QCd; ++k)
            s += bf2f(sh_qc[rq * 2048 + row * 128 + qb + k]) *
                 bf2f(sh_kc[rs * 2048 + row * 128 + qb + k]);
          sc[rq][rs] = s * inv_sq;
        }
#pragma unroll 1
      for (int rq = 0; rq < RR; ++rq) {
        float mx = sc[rq][0];
#pragma unroll
        for (int rs = 1; rs < RR; ++rs) mx = fmaxf(mx, sc[rq][rs]);
        float p[RR], den = 0.0f;
#pragma unroll
        for (int rs = 0; rs < RR; ++rs) { p[rs] = __expf(sc[rq][rs] - mx); den += p[rs]; }
        float inv = 1.0f / den;
#pragma unroll 1
        for (int vcn = 0; vcn < VCd; ++vcn) {
          float o = 0.0f;
#pragma unroll
          for (int rs = 0; rs < RR; ++rs)
            o += p[rs] * sh_vc[rs * 1024 + row * 64 + vb + vcn];
          sh_q[rq * 1024 + row * 64 + vb + vcn] = o * inv;  // reuse sh_q as comm
        }
      }
    }
    __syncthreads();

    // ---- H: gated state update (active rims get h_new + comm) ----------
#pragma unroll 1
    for (int i = lane; i < 16 * HH; i += 32) {
      int row = i >> 6, col = i & 63;
      float m  = sh_am[row * 8 + wave];
      float hn = sh_hn[wave * 1024 + i] + sh_q[wave * 1024 + i];
      float ho = sh_h[wave * 1024 + i];
      float hv = m * hn + (1.0f - m) * ho;
      sh_h[wave * 1024 + i] = hv;
      sh_hb[wave * 1024 + apack_idx(row, col)] = f2bf(hv);
    }
    __syncthreads();
  }

  // ---- MLP head: relu(concat(h, statics) @ W1 + b1) @ W2 + b2 ----------
  if (tid < 16) {
    int row = tid, b = b0 + row;
    float hid[10];
#pragma unroll 1
    for (int j = 0; j < 10; ++j) {
      float s = b1[j];
#pragma unroll 1
      for (int r = 0; r < RR; ++r)
#pragma unroll 8
        for (int hc = 0; hc < HH; ++hc)
          s += sh_h[r * 1024 + row * HH + hc] * W1[(r * HH + hc) * 10 + j];
#pragma unroll
      for (int ss = 0; ss < SSd; ++ss)
        s += statics[b * SSd + ss] * W1[(RR * HH + ss) * 10 + j];
      hid[j] = fmaxf(s, 0.0f);
    }
    float o0 = b2[0], o1 = b2[1];
#pragma unroll
    for (int j = 0; j < 10; ++j) { o0 += hid[j] * W2[j * 2]; o1 += hid[j] * W2[j * 2 + 1]; }
    out[b * 2 + 0] = o0;
    out[b * 2 + 1] = o1;
  }
}

// ---------------- host launcher ----------------

extern "C" void kernel_launch(void* const* d_in, const int* in_sizes, int n_in,
                              void* d_out, int out_size, void* d_ws, size_t ws_size,
                              hipStream_t stream) {
  (void)in_sizes; (void)n_in; (void)out_size; (void)ws_size;
  const float* x       = (const float*)d_in[0];
  const float* statics = (const float*)d_in[1];
  const float* maskp   = (const float*)d_in[2];
  const float* delta   = (const float*)d_in[3];
  const float* xlast   = (const float*)d_in[4];
  const float* xmean   = (const float*)d_in[5];
  const float* wdg     = (const float*)d_in[6];
  const float* bdg     = (const float*)d_in[7];
  const float* Wk      = (const float*)d_in[8];
  const float* Wv      = (const float*)d_in[9];
  const float* Wq      = (const float*)d_in[10];
  const float* W_ih    = (const float*)d_in[11];
  const float* W_hh    = (const float*)d_in[12];
  const float* b_ih    = (const float*)d_in[13];
  const float* b_hh    = (const float*)d_in[14];
  const float* Wq_c    = (const float*)d_in[15];
  const float* Wk_c    = (const float*)d_in[16];
  const float* Wv_c    = (const float*)d_in[17];
  const float* W1      = (const float*)d_in[18];
  const float* b1      = (const float*)d_in[19];
  const float* W2      = (const float*)d_in[20];
  const float* b2      = (const float*)d_in[21];
  float* out = (float*)d_out;
  __bf16* ws = (__bf16*)d_ws;

  // Pack weights into WMMA B-fragment layout (bf16) — cheap, L2-resident.
  pack_weights<<<64,  256, 0, stream>>>(Wk,   ws + OFF_WK,  1,  64,  64);
  pack_weights<<<64,  256, 0, stream>>>(Wv,   ws + OFF_WV,  1,  64, 128);
  pack_weights<<<128, 256, 0, stream>>>(Wq,   ws + OFF_WQ,  6,  64,  64);
  pack_weights<<<576, 256, 0, stream>>>(W_ih, ws + OFF_WIH, 6, 128, 192);
  pack_weights<<<288, 256, 0, stream>>>(W_hh, ws + OFF_WHH, 6,  64, 192);
  pack_weights<<<192, 256, 0, stream>>>(Wq_c, ws + OFF_WQC, 6,  64, 128);
  pack_weights<<<192, 256, 0, stream>>>(Wk_c, ws + OFF_WKC, 6,  64, 128);
  pack_weights<<<128, 256, 0, stream>>>(Wv_c, ws + OFF_WVC, 6,  64,  64);

  // Allow >64KB dynamic LDS (WGP has 320KB); attribute set is not a stream op.
  (void)hipFuncSetAttribute((const void*)rims_recurrence_kernel,
                            hipFuncAttributeMaxDynamicSharedMemorySize,
                            SMEM_BYTES);

  rims_recurrence_kernel<<<BB / 16, NTHREADS, SMEM_BYTES, stream>>>(
      x, statics, maskp, delta, xlast, xmean, wdg, bdg,
      b_ih, b_hh, W1, b1, W2, b2, ws, out);
}
